// MoEFFN_35570919145599
// MI455X (gfx1250) — compile-verified, matrix operand order checked by reading, not run on hardware
//
#include <hip/hip_runtime.h>
#include <stdint.h>
#include <math.h>

static constexpr int kE   = 8;      // experts
static constexpr int kC   = 1024;   // DIM
static constexpr int kH   = 2048;   // HID
static constexpr int kN   = 4096;   // tokens (2*2048)
static constexpr int kCAP = 9216;   // 2N + E*128 aligned entry capacity
static constexpr int TM   = 128;    // block tile M (entries)
static constexpr int TN   = 128;    // block tile N
static constexpr int TK   = 32;     // K step (bf16 WMMA K)

typedef __attribute__((ext_vector_type(16))) __bf16 v16bf;
typedef __attribute__((ext_vector_type(8)))  float  v8f;
typedef __attribute__((ext_vector_type(4)))  unsigned int v4u;
typedef __attribute__((ext_vector_type(8)))  int v8i;
typedef __attribute__((ext_vector_type(4)))  int v4i;

__device__ __forceinline__ unsigned short f2bf(float f) {
  union { float f; unsigned int u; } v; v.f = f;
  unsigned int u = v.u;
  unsigned int r = (u + 0x7FFFu + ((u >> 16) & 1u)) >> 16;  // RNE
  return (unsigned short)r;
}
__device__ __forceinline__ float bf2f(unsigned short h) {
  union { unsigned int u; float f; } v; v.u = ((unsigned int)h) << 16;
  return v.f;
}
__device__ __forceinline__ unsigned lds_off(const void* p) {
  // generic LDS pointer: aperture in high 32 bits, LDS byte offset in low 32
  return (unsigned)(uintptr_t)p;
}

// 2D tile DMA: global (bf16, row-major, row length = stride0 elems) -> packed LDS.
// tile0 = contiguous width (elems), tile1 = rows. Issued once per executing wave.
__device__ __forceinline__ void tdm_load_2d(unsigned ldsByteOff, const void* gptr,
                                            unsigned td0, unsigned td1,
                                            unsigned tile0, unsigned tile1,
                                            unsigned stride0) {
  unsigned long long ga = (unsigned long long)(uintptr_t)gptr;
  v4u g0;
  g0[0] = 1u;                                   // count=1 (valid), load, user mode
  g0[1] = ldsByteOff;                           // lds_addr
  g0[2] = (unsigned)(ga & 0xffffffffu);         // global_addr[31:0]
  g0[3] = (unsigned)((ga >> 32) & 0x1ffffffu)   // global_addr[56:32]
          | (2u << 30);                         // type = 2 ("image")
  v8i g1;
  g1[0] = (int)(1u << 16);                      // workgroup_mask=0, data_size=1 (2B)
  g1[1] = (int)((td0 & 0xffffu) << 16);         // tensor_dim0[15:0] @ [63:48]
  g1[2] = (int)(((td0 >> 16) & 0xffffu) | ((td1 & 0xffffu) << 16));
  g1[3] = (int)(((td1 >> 16) & 0xffffu) | ((tile0 & 0xffffu) << 16));
  g1[4] = (int)(tile1 & 0xffffu);               // tile_dim1, tile_dim2=0
  g1[5] = (int)stride0;                         // tensor_dim0_stride[31:0]
  g1[6] = 0;
  g1[7] = 0;
  v4i g2 = {0, 0, 0, 0};
  v4i g3 = {0, 0, 0, 0};
#if defined(__clang_major__) && (__clang_major__ >= 23)
  v8i g4 = {0, 0, 0, 0, 0, 0, 0, 0};
  __builtin_amdgcn_tensor_load_to_lds(g0, g1, g2, g3, g4, 0);
#else
  __builtin_amdgcn_tensor_load_to_lds(g0, g1, g2, g3, 0);
#endif
}

// ---------------- setup kernels ----------------

__global__ void k_init(int* gtok, int* counts, int* cursors) {
  int i = blockIdx.x * blockDim.x + threadIdx.x;
  if (i < kCAP) gtok[i] = -1;
  if (i < kE) { counts[i] = 0; cursors[i] = 0; }
}

__global__ void k_gate(const float* __restrict__ x, const float* __restrict__ Wg,
                       int* __restrict__ topi, float* __restrict__ topw,
                       int* __restrict__ counts) {
  int n = blockIdx.x * blockDim.x + threadIdx.x;
  if (n >= kN) return;
  float acc[kE];
#pragma unroll
  for (int e = 0; e < kE; e++) acc[e] = 0.f;
  const float* xr = x + (size_t)n * kC;
  for (int d = 0; d < kC; d++) {
    float xv = xr[d];
    const float* wr = Wg + (size_t)d * kE;
#pragma unroll
    for (int e = 0; e < kE; e++) acc[e] += xv * wr[e];
  }
  int e0 = 0; float l0 = acc[0];
#pragma unroll
  for (int e = 1; e < kE; e++) if (acc[e] > l0) { l0 = acc[e]; e0 = e; }
  int e1 = -1; float l1 = -3.4e38f;
#pragma unroll
  for (int e = 0; e < kE; e++) if (e != e0 && acc[e] > l1) { l1 = acc[e]; e1 = e; }
  float s1  = __expf(l1 - l0);
  float inv = 1.f / (1.f + s1);
  topi[2 * n]     = e0;  topi[2 * n + 1] = e1;
  topw[2 * n]     = inv; topw[2 * n + 1] = s1 * inv;
  atomicAdd(&counts[e0], 1);
  atomicAdd(&counts[e1], 1);
}

__global__ void k_scan(const int* __restrict__ counts, int* __restrict__ aoff) {
  if (blockIdx.x == 0 && threadIdx.x == 0) {
    int o = 0;
    for (int e = 0; e < kE; e++) {
      aoff[e] = o;
      o += (counts[e] + TM - 1) & ~(TM - 1);
    }
    aoff[kE] = o;
  }
}

__global__ void k_scatter(const int* __restrict__ topi, const int* __restrict__ aoff,
                          int* __restrict__ cursors, int* __restrict__ gtok,
                          int* __restrict__ gslot) {
  int n = blockIdx.x * blockDim.x + threadIdx.x;
  if (n >= kN) return;
#pragma unroll
  for (int k = 0; k < 2; k++) {
    int e = topi[2 * n + k];
    int pos = atomicAdd(&cursors[e], 1);
    int ent = aoff[e] + pos;
    gtok[ent]  = n;
    gslot[ent] = 2 * n + k;
  }
}

__global__ void k_xconv(const float* __restrict__ x, unsigned int* __restrict__ xb2) {
  int i = blockIdx.x * blockDim.x + threadIdx.x;  // bf16 pairs
  if (i >= kN * kC / 2) return;
  float a = x[2 * i], b = x[2 * i + 1];
  xb2[i] = (unsigned int)f2bf(a) | ((unsigned int)f2bf(b) << 16);
}

// Transpose + convert: W [E][R][Cc] f32 -> Wt [E][Cc][R] bf16. Block (32,8).
__global__ void k_wtc(const float* __restrict__ W, unsigned short* __restrict__ Wt,
                      int R, int Cc) {
  __shared__ float tile[32][33];
  int e  = blockIdx.z;
  int c0 = blockIdx.x * 32;   // along Cc
  int r0 = blockIdx.y * 32;   // along R
  int tx = threadIdx.x, ty = threadIdx.y;
  const float* Wb = W + (size_t)e * R * Cc;
  unsigned short* Wtb = Wt + (size_t)e * R * Cc;
#pragma unroll
  for (int i = 0; i < 32; i += 8)
    tile[ty + i][tx] = Wb[(size_t)(r0 + ty + i) * Cc + c0 + tx];
  __syncthreads();
#pragma unroll
  for (int i = 0; i < 32; i += 8)
    Wtb[(size_t)(c0 + ty + i) * R + r0 + tx] = f2bf(tile[tx][ty + i]);
}

// ---------------- phase 1: T = silu(x@W1[e]) * (x@W2[e]) ----------------
// B tiles (from transposed bf16 weights) streamed by TDM, double buffered.

__global__ __launch_bounds__(256)
void k_ffn1(const unsigned short* __restrict__ xb,
            const unsigned short* __restrict__ W1t,
            const unsigned short* __restrict__ W2t,
            const int* __restrict__ gtok, const int* __restrict__ aoff,
            unsigned short* __restrict__ T) {
  __shared__ alignas(32) unsigned short Asm[TM * TK];
  __shared__ alignas(32) unsigned short B1s[2][TN * TK];
  __shared__ alignas(32) unsigned short B2s[2][TN * TK];
  __shared__ int toks[TM];

  const int mBase = blockIdx.y * TM;
  const int hBase = blockIdx.x * TN;
  const int tid   = threadIdx.x;

  if (mBase >= aoff[kE]) return;  // beyond used capacity (uniform)
  int e = 0;
#pragma unroll
  for (int i = 0; i < kE; i++) if (mBase >= aoff[i + 1]) e = i + 1;

  if (tid < TM) toks[tid] = gtok[mBase + tid];
  __syncthreads();
  if (toks[0] < 0) return;  // whole tile is segment padding (uniform)

  const int wave = tid >> 5, lane = tid & 31;
  const int wm = wave & 3;        // m offset wm*32
  const int wn = wave >> 2;       // n offset wn*64
  const int lrow = lane & 15, lhalf = lane >> 4;

  v8f acc1[2][4], acc2[2][4];
#pragma unroll
  for (int i = 0; i < 2; i++)
#pragma unroll
    for (int j = 0; j < 4; j++)
#pragma unroll
      for (int t = 0; t < 8; t++) { acc1[i][j][t] = 0.f; acc2[i][j][t] = 0.f; }

  // W1t/W2t: [E][kH][kC] bf16; tile rows = h in [hBase, hBase+TN), cols = k
  const unsigned short* Wb1 = W1t + ((size_t)e * kH + hBase) * kC;
  const unsigned short* Wb2 = W2t + ((size_t)e * kH + hBase) * kC;

  if (wave == 0) {  // prologue: DMA first B tiles into buffer 0
    tdm_load_2d(lds_off(&B1s[0][0]), Wb1, kC, kH, TK, TN, kC);
    tdm_load_2d(lds_off(&B2s[0][0]), Wb2, kC, kH, TK, TN, kC);
  }

  int ib = 0;
  for (int kb = 0; kb < kC; kb += TK, ib ^= 1) {
    __syncthreads();  // prev compute done: safe to overwrite A and B[ib^1]
    if (wave == 0 && kb + TK < kC) {  // DMA next B tiles into other buffer
      tdm_load_2d(lds_off(&B1s[ib ^ 1][0]), Wb1 + kb + TK, kC, kH, TK, TN, kC);
      tdm_load_2d(lds_off(&B2s[ib ^ 1][0]), Wb2 + kb + TK, kC, kH, TK, TN, kC);
    }
    // stage A (gathered token rows), 128x32 bf16 as 2048 dwords
#pragma unroll
    for (int it = 0; it < 8; it++) {
      int idx = tid + it * 256;          // 0..2047
      int m = idx >> 4, j = idx & 15;
      int tok = toks[m];
      unsigned int val = 0;
      if (tok >= 0) {
        const unsigned int* p = (const unsigned int*)(xb + (size_t)tok * kC + kb);
        val = p[j];
      }
      ((unsigned int*)Asm)[idx] = val;
    }
    if (wave == 0) {  // retire oldest TDM pair (in-order completion)
      if (kb + TK < kC) __builtin_amdgcn_s_wait_tensorcnt(2);
      else              __builtin_amdgcn_s_wait_tensorcnt(0);
    }
    __syncthreads();

    v16bf a[2];
#pragma unroll
    for (int i = 0; i < 2; i++) {
      int r = wm * 32 + i * 16 + lrow;
      a[i] = *(const v16bf*)(Asm + r * TK + lhalf * 16);
    }
#pragma unroll
    for (int j = 0; j < 4; j++) {
      int cN = wn * 64 + j * 16 + lrow;
      v16bf b1 = *(const v16bf*)(&B1s[ib][0] + cN * TK + lhalf * 16);
      v16bf b2 = *(const v16bf*)(&B2s[ib][0] + cN * TK + lhalf * 16);
#pragma unroll
      for (int i = 0; i < 2; i++) {
        acc1[i][j] = __builtin_amdgcn_wmma_f32_16x16x32_bf16(
            false, a[i], false, b1, (short)0, acc1[i][j], false, false);
        acc2[i][j] = __builtin_amdgcn_wmma_f32_16x16x32_bf16(
            false, a[i], false, b2, (short)0, acc2[i][j], false, false);
      }
    }
  }

  // epilogue: silu(h1)*h2 -> T (bf16). C/D layout: VGPR v holds M=v (+8 hi lanes), N=lane%16.
#pragma unroll
  for (int i = 0; i < 2; i++)
#pragma unroll
    for (int j = 0; j < 4; j++)
#pragma unroll
      for (int v = 0; v < 8; v++) {
        int mloc = wm * 32 + i * 16 + lhalf * 8 + v;
        if (toks[mloc] < 0) continue;
        int col = hBase + wn * 64 + j * 16 + lrow;
        float h1 = acc1[i][j][v];
        float h2 = acc2[i][j][v];
        float s  = h1 / (1.f + __expf(-h1));
        T[(size_t)(mBase + mloc) * kH + col] = f2bf(s * h2);
      }
}

// ---------------- phase 2: Y[slot] = T @ W3[e] ----------------

__global__ __launch_bounds__(256)
void k_ffn2(const unsigned short* __restrict__ T,
            const unsigned short* __restrict__ W3t,
            const int* __restrict__ gtok, const int* __restrict__ gslot,
            const int* __restrict__ aoff, unsigned short* __restrict__ Y) {
  __shared__ alignas(32) unsigned short Asm[TM * TK];
  __shared__ alignas(32) unsigned short Bs[2][TN * TK];
  __shared__ int slots[TM];

  const int mBase = blockIdx.y * TM;
  const int cBase = blockIdx.x * TN;
  const int tid   = threadIdx.x;

  if (mBase >= aoff[kE]) return;
  int e = 0;
#pragma unroll
  for (int i = 0; i < kE; i++) if (mBase >= aoff[i + 1]) e = i + 1;

  if (tid < TM) slots[tid] = (gtok[mBase + tid] >= 0) ? gslot[mBase + tid] : -1;
  __syncthreads();
  if (slots[0] < 0) return;

  const int wave = tid >> 5, lane = tid & 31;
  const int wm = wave & 3, wn = wave >> 2;
  const int lrow = lane & 15, lhalf = lane >> 4;

  v8f acc[2][4];
#pragma unroll
  for (int i = 0; i < 2; i++)
#pragma unroll
    for (int j = 0; j < 4; j++)
#pragma unroll
      for (int t = 0; t < 8; t++) acc[i][j][t] = 0.f;

  // W3t: [E][kC][kH] bf16; tile rows = c in [cBase, cBase+TN), cols = k (H dim)
  const unsigned short* Wb3 = W3t + ((size_t)e * kC + cBase) * kH;

  if (wave == 0) {
    tdm_load_2d(lds_off(&Bs[0][0]), Wb3, kH, kC, TK, TN, kH);
  }

  int ib = 0;
  for (int kb = 0; kb < kH; kb += TK, ib ^= 1) {
    __syncthreads();
    if (wave == 0 && kb + TK < kH)
      tdm_load_2d(lds_off(&Bs[ib ^ 1][0]), Wb3 + kb + TK, kH, kC, TK, TN, kH);
    // stage A from T (entry rows are contiguous)
#pragma unroll
    for (int it = 0; it < 8; it++) {
      int idx = tid + it * 256;
      int m = idx >> 4, j = idx & 15;
      const unsigned int* p = (const unsigned int*)(T + (size_t)(mBase + m) * kH + kb);
      ((unsigned int*)Asm)[idx] = p[j];
    }
    if (wave == 0) {
      if (kb + TK < kH) __builtin_amdgcn_s_wait_tensorcnt(1);
      else              __builtin_amdgcn_s_wait_tensorcnt(0);
    }
    __syncthreads();

    v16bf a[2];
#pragma unroll
    for (int i = 0; i < 2; i++) {
      int r = wm * 32 + i * 16 + lrow;
      a[i] = *(const v16bf*)(Asm + r * TK + lhalf * 16);
    }
#pragma unroll
    for (int j = 0; j < 4; j++) {
      int cN = wn * 64 + j * 16 + lrow;
      v16bf b = *(const v16bf*)(&Bs[ib][0] + cN * TK + lhalf * 16);
#pragma unroll
      for (int i = 0; i < 2; i++)
        acc[i][j] = __builtin_amdgcn_wmma_f32_16x16x32_bf16(
            false, a[i], false, b, (short)0, acc[i][j], false, false);
    }
  }

#pragma unroll
  for (int i = 0; i < 2; i++)
#pragma unroll
    for (int j = 0; j < 4; j++)
#pragma unroll
      for (int v = 0; v < 8; v++) {
        int mloc = wm * 32 + i * 16 + lhalf * 8 + v;
        int slot = slots[mloc];
        if (slot < 0) continue;
        int col = cBase + wn * 64 + j * 16 + lrow;
        Y[(size_t)slot * kC + col] = f2bf(acc[i][j][v]);
      }
}

// ---------------- final combine (fixed order -> deterministic) ----------------

__global__ void k_combine(const unsigned short* __restrict__ Y,
                          const float* __restrict__ topw, float* __restrict__ out) {
  int i = blockIdx.x * blockDim.x + threadIdx.x;
  if (i >= kN * kC) return;
  int n = i >> 10;          // /kC
  int c = i & (kC - 1);
  float w0 = topw[2 * n], w1 = topw[2 * n + 1];
  float y0 = bf2f(Y[(size_t)(2 * n) * kC + c]);
  float y1 = bf2f(Y[(size_t)(2 * n + 1) * kC + c]);
  out[i] = w0 * y0 + w1 * y1;
}

// ---------------- launch ----------------

extern "C" void kernel_launch(void* const* d_in, const int* in_sizes, int n_in,
                              void* d_out, int out_size, void* d_ws, size_t ws_size,
                              hipStream_t stream) {
  (void)in_sizes; (void)n_in; (void)out_size; (void)ws_size;
  const float* x  = (const float*)d_in[0];
  const float* Wg = (const float*)d_in[1];
  const float* W1 = (const float*)d_in[2];
  const float* W2 = (const float*)d_in[3];
  const float* W3 = (const float*)d_in[4];
  float* out = (float*)d_out;

  char* ws = (char*)d_ws;
  size_t o = 0;
  auto alloc = [&](size_t bytes) -> char* {
    char* p = ws + o;
    o += (bytes + 255) & ~(size_t)255;
    return p;
  };
  int*   topi    = (int*)alloc((size_t)kN * 2 * 4);
  float* topw    = (float*)alloc((size_t)kN * 2 * 4);
  int*   counts  = (int*)alloc(kE * 4);
  int*   cursors = (int*)alloc(kE * 4);
  int*   aoff    = (int*)alloc((kE + 1) * 4);
  int*   gtok    = (int*)alloc((size_t)kCAP * 4);
  int*   gslot   = (int*)alloc((size_t)kCAP * 4);
  unsigned short* xb  = (unsigned short*)alloc((size_t)kN * kC * 2);
  unsigned short* T   = (unsigned short*)alloc((size_t)kCAP * kH * 2);
  unsigned short* Y   = (unsigned short*)alloc((size_t)kN * 2 * kC * 2);
  unsigned short* W1t = (unsigned short*)alloc((size_t)kE * kC * kH * 2);
  unsigned short* W2t = (unsigned short*)alloc((size_t)kE * kC * kH * 2);
  unsigned short* W3t = (unsigned short*)alloc((size_t)kE * kH * kC * 2);

  k_init<<<(kCAP + 255) / 256, 256, 0, stream>>>(gtok, counts, cursors);
  k_gate<<<(kN + 255) / 256, 256, 0, stream>>>(x, Wg, topi, topw, counts);
  k_scan<<<1, 32, 0, stream>>>(counts, aoff);
  k_scatter<<<(kN + 255) / 256, 256, 0, stream>>>(topi, aoff, cursors, gtok, gslot);
  k_xconv<<<(kN * kC / 2 + 255) / 256, 256, 0, stream>>>(x, (unsigned int*)xb);

  dim3 bt(32, 8);
  k_wtc<<<dim3(kH / 32, kC / 32, kE), bt, 0, stream>>>(W1, W1t, kC, kH);  // -> [E][H][C]
  k_wtc<<<dim3(kH / 32, kC / 32, kE), bt, 0, stream>>>(W2, W2t, kC, kH);  // -> [E][H][C]
  k_wtc<<<dim3(kC / 32, kH / 32, kE), bt, 0, stream>>>(W3, W3t, kH, kC);  // -> [E][C][H]

  dim3 g1(kH / TN, kCAP / TM);   // 16 x 72
  k_ffn1<<<g1, 256, 0, stream>>>(xb, W1t, W2t, gtok, aoff, T);

  dim3 g2(kC / TN, kCAP / TM);   // 8 x 72
  k_ffn2<<<g2, 256, 0, stream>>>(T, W3t, gtok, gslot, aoff, Y);

  k_combine<<<(kN * kC + 255) / 256, 256, 0, stream>>>(Y, topw, out);
}
